// HierarchicalGNN_79035988180996
// MI455X (gfx1250) — matrix operand order repaired and checked by reading, not compiled, not run on hardware
//
#include <hip/hip_runtime.h>

// ---------------------------------------------------------------------------
// CDNA5 (gfx1250) wave32 bf16 WMMA implementation of the hierarchical GNN cell
// B=256, E=128, N=128, T=20, D=128
// ---------------------------------------------------------------------------

typedef __bf16 bf16;
typedef __attribute__((ext_vector_type(16))) __bf16 v16bf;
typedef __attribute__((ext_vector_type(8)))  __bf16 v8bf;
typedef __attribute__((ext_vector_type(8)))  float  v8f;

#define DEV static __device__ __forceinline__

DEV bf16 f2bf(float f) {
  unsigned u = __builtin_bit_cast(unsigned, f);
  u += 0x7FFFu + ((u >> 16) & 1u);            // round-to-nearest-even
  unsigned short s = (unsigned short)(u >> 16);
  return __builtin_bit_cast(bf16, s);
}

union Frag { v16bf v; v8bf h[2]; };

// A-matrix 16x32 bf16 fragment from LDS rows [16][rowstride], row-major.
// Lane l holds M=l%16; halves 0..7 -> K = kk + (l>=16?8:0) + 0..7,
//                      halves 8..15 -> K = kk+16 + (l>=16?8:0) + 0..7.
DEV v16bf load_a_frag(const bf16* rows, int rowstride, int kk, int lane) {
  const int m   = lane & 15;
  const int off = (lane >= 16) ? 8 : 0;
  const bf16* p = rows + m * rowstride + kk + off;
  Frag f;
  f.h[0] = *(const v8bf*)(p);
  f.h[1] = *(const v8bf*)(p + 16);
  return f.v;
}

// B-matrix 32x16 bf16 fragment. Weight stored [N][K] row-major in LDS.
// Lane l holds N=col0+l%16; halves 0..7 -> K = kk + (l>=16?8:0) + 0..7,
//                          halves 8..15 -> K = kk+16 + (l>=16?8:0) + 0..7.
DEV v16bf load_b_frag(const bf16* w, int rowstride, int col0, int kk, int lane) {
  const int n    = col0 + (lane & 15);
  const int koff = kk + ((lane >= 16) ? 8 : 0);
  const bf16* p  = w + n * rowstride + koff;
  Frag f;
  f.h[0] = *(const v8bf*)(p);
  f.h[1] = *(const v8bf*)(p + 16);
  return f.v;
}

DEV v8f wmma_bf16(v16bf a, v16bf b, v8f c) {
  return __builtin_amdgcn_wmma_f32_16x16x32_bf16(
      /*neg_a=*/false, a, /*neg_b=*/false, b,
      /*c_mod=*/(short)0, c, /*reuse_a=*/false, /*reuse_b=*/false);
}

DEV v8f zero8() { v8f z = {0.f,0.f,0.f,0.f,0.f,0.f,0.f,0.f}; return z; }

// Fast gate nonlinearities using single-instruction CDNA5 transcendentals
// (v_exp_f32 + v_rcp_f32) -- avoids the v_div_scale/v_div_fmas IEEE divide
// expansion on the serial GRU critical path.
#define LOG2E 1.442695040888963f
DEV float fast_sigmoid(float x) {
  return __builtin_amdgcn_rcpf(1.0f + __builtin_amdgcn_exp2f(-LOG2E * x));
}
DEV float fast_tanh(float x) {
  // tanh(x) = 1 - 2/(exp(2x)+1)
  float e = __builtin_amdgcn_exp2f((2.0f * LOG2E) * x);
  return 1.0f - 2.0f * __builtin_amdgcn_rcpf(e + 1.0f);
}

// ---------------------------------------------------------------------------
// kernel 0: f32 -> bf16 weight conversion (run once; consumers hit L2)
// ---------------------------------------------------------------------------
__global__ void cvt_bf16_kernel(const float* __restrict__ src,
                                bf16* __restrict__ dst, int n) {
  int i = blockIdx.x * blockDim.x + threadIdx.x;
  if (i < n) dst[i] = f2bf(src[i]);
}

// ---------------------------------------------------------------------------
// kernel 1: fused 20-step GRU over 32768 sequences. 16 seqs/block, 8 waves.
// Wave w owns D-columns [16w,16w+16) and keeps ALL 24 weight B-fragments
// resident in VGPRs across the whole t-loop (weights are step-invariant),
// cutting per-step LDS traffic 4x. x_{t+1} is prefetched into registers
// during step t's WMMAs (double-buffered LDS) to hide HBM latency of the
// 335MB micro_actions stream.
// LDS: Wih 96KB | Whh 96KB | x_t 2x4KB | h 4KB = 204KB  (1 WG / WGP)
// ---------------------------------------------------------------------------
__global__ __launch_bounds__(256) void gru_kernel(
    const float* __restrict__ x, const int* __restrict__ lens_g,
    const bf16* __restrict__ wih_g, const bf16* __restrict__ whh_g,
    const float* __restrict__ bih, const float* __restrict__ bhh,
    bf16* __restrict__ mh)
{
  extern __shared__ bf16 smem[];
  bf16* Wih = smem;             // [384][128]
  bf16* Whh = Wih + 49152;      // [384][128]
  bf16* X0  = Whh + 49152;      // [16][128] double-buffered x_t
  bf16* X1  = X0 + 2048;
  bf16* Hl  = X1 + 2048;        // [16][128]

  const int tid  = threadIdx.x;
  const int lane = tid & 31;
  const int wave = tid >> 5;
  const int seq0 = blockIdx.x << 4;

  {
    const int4* s0 = (const int4*)wih_g;
    const int4* s1 = (const int4*)whh_g;
    int4* d0 = (int4*)Wih; int4* d1 = (int4*)Whh;
    for (int i = tid; i < 49152 / 8; i += 256) { d0[i] = s0[i]; d1[i] = s1[i]; }
  }
  for (int i = tid; i < 2048; i += 256) Hl[i] = f2bf(0.0f);

  // per-thread x prefetch slot: row pm (0..15), cols pk..pk+7
  const int pm = tid >> 4;
  const int pk = (tid & 15) << 3;
  const float* xbase = x + (size_t)(seq0 + pm) * (20 * 128) + pk;

  // stage x_0 into X0
  float4 fa = ((const float4*)xbase)[0];
  float4 fb = ((const float4*)xbase)[1];
  {
    union { bf16 b[8]; int4 q; } u;
    u.b[0] = f2bf(fa.x); u.b[1] = f2bf(fa.y); u.b[2] = f2bf(fa.z); u.b[3] = f2bf(fa.w);
    u.b[4] = f2bf(fb.x); u.b[5] = f2bf(fb.y); u.b[6] = f2bf(fb.z); u.b[7] = f2bf(fb.w);
    *(int4*)(X0 + pm * 128 + pk) = u.q;
  }
  __syncthreads();   // weights staged: safe to hoist B-fragments

  // hoist step-invariant weight fragments into registers:
  // [gate g=0..2][kk tile q=0..3]
  v16bf bi_f[12], bh_f[12];
  #pragma unroll
  for (int g = 0; g < 3; ++g)
    #pragma unroll
    for (int q = 0; q < 4; ++q) {
      bi_f[g * 4 + q] = load_b_frag(Wih, 128, g * 128 + (wave << 4), q * 32, lane);
      bh_f[g * 4 + q] = load_b_frag(Whh, 128, g * 128 + (wave << 4), q * 32, lane);
    }

  const int col = (wave << 4) + (lane & 15);
  const int mb  = (lane >= 16) ? 8 : 0;
  const float bir = bih[col],       bhr = bhh[col];
  const float biz = bih[128 + col], bhz = bhh[128 + col];
  const float bin = bih[256 + col], bhn = bhh[256 + col];

  int   len[8];
  float h[8];
  #pragma unroll
  for (int r = 0; r < 8; ++r) { len[r] = lens_g[seq0 + mb + r]; h[r] = 0.0f; }

  for (int t = 0; t < 20; ++t) {
    bf16* Xc = (t & 1) ? X1 : X0;
    bf16* Xn = (t & 1) ? X0 : X1;
    __syncthreads();            // B1: Xc (stored at t-1) + Hl writes visible

    if (t + 1 < 20) {           // prefetch x_{t+1}; latency hidden by WMMAs
      const float4* p = (const float4*)(xbase + (size_t)(t + 1) * 128);
      fa = p[0]; fb = p[1];
    }

    v8f ir = zero8(), iz = zero8(), in_ = zero8();
    v8f hr = zero8(), hz = zero8(), hn = zero8();
    #pragma unroll
    for (int q = 0; q < 4; ++q) {
      v16bf ax = load_a_frag(Xc, 128, q * 32, lane);
      v16bf ah = load_a_frag(Hl, 128, q * 32, lane);
      ir  = wmma_bf16(ax, bi_f[q],     ir);
      iz  = wmma_bf16(ax, bi_f[4 + q], iz);
      in_ = wmma_bf16(ax, bi_f[8 + q], in_);
      hr  = wmma_bf16(ah, bh_f[q],     hr);
      hz  = wmma_bf16(ah, bh_f[4 + q], hz);
      hn  = wmma_bf16(ah, bh_f[8 + q], hn);
    }
    __syncthreads();            // B2: all reads of Xc / Hl done

    #pragma unroll
    for (int r = 0; r < 8; ++r) {
      float rg = fast_sigmoid(ir[r] + bir + hr[r] + bhr);
      float zg = fast_sigmoid(iz[r] + biz + hz[r] + bhz);
      float ng = fast_tanh(in_[r] + bin + rg * (hn[r] + bhn));
      float hu = (1.0f - zg) * ng + zg * h[r];
      if (t < len[r]) h[r] = hu;               // freeze past true length
      Hl[(mb + r) * 128 + col] = f2bf(h[r]);
    }

    if (t + 1 < 20) {           // stage prefetched x_{t+1} into other buffer
      union { bf16 b[8]; int4 q; } u;
      u.b[0] = f2bf(fa.x); u.b[1] = f2bf(fa.y); u.b[2] = f2bf(fa.z); u.b[3] = f2bf(fa.w);
      u.b[4] = f2bf(fb.x); u.b[5] = f2bf(fb.y); u.b[6] = f2bf(fb.z); u.b[7] = f2bf(fb.w);
      *(int4*)(Xn + pm * 128 + pk) = u.q;
    }
  }
  #pragma unroll
  for (int r = 0; r < 8; ++r)
    mh[(size_t)(seq0 + mb + r) * 128 + col] = f2bf(h[r]);
}

// ---------------------------------------------------------------------------
// kernel 2: h_in/h_out = [macro_items | mh] @ {w_in,w_out}.T + bias
// Output written TRANSPOSED per batch ([B][D][E]) so kernel 3's B-matrix
// fragments read contiguous K from LDS.
// LDS: Win 64KB | Wout 64KB | A rows 8KB = 136KB
// ---------------------------------------------------------------------------
__global__ __launch_bounds__(256) void inout_kernel(
    const float* __restrict__ macro_items, const bf16* __restrict__ mh,
    const bf16* __restrict__ win_g, const bf16* __restrict__ wout_g,
    const float* __restrict__ b_in, const float* __restrict__ b_out,
    bf16* __restrict__ hin_t, bf16* __restrict__ hout_t)
{
  extern __shared__ bf16 smem[];
  bf16* Win  = smem;            // [128][256]
  bf16* Wout = Win + 32768;     // [128][256]
  bf16* Ar   = Wout + 32768;    // [16][256]

  const int tid  = threadIdx.x;
  const int lane = tid & 31;
  const int wave = tid >> 5;
  const int seq0 = blockIdx.x << 4;

  {
    const int4* s0 = (const int4*)win_g;
    const int4* s1 = (const int4*)wout_g;
    int4* d0 = (int4*)Win; int4* d1 = (int4*)Wout;
    for (int i = tid; i < 32768 / 8; i += 256) { d0[i] = s0[i]; d1[i] = s1[i]; }
  }
  for (int i = tid; i < 2048; i += 256) {
    int m = i >> 7, k = i & 127;
    Ar[m * 256 + k]       = f2bf(macro_items[(size_t)(seq0 + m) * 128 + k]);
    Ar[m * 256 + 128 + k] = mh[(size_t)(seq0 + m) * 128 + k];
  }
  __syncthreads();

  v8f ci = zero8(), co = zero8();
  #pragma unroll
  for (int kk = 0; kk < 256; kk += 32) {
    v16bf a = load_a_frag(Ar, 256, kk, lane);
    ci = wmma_bf16(a, load_b_frag(Win,  256, wave << 4, kk, lane), ci);
    co = wmma_bf16(a, load_b_frag(Wout, 256, wave << 4, kk, lane), co);
  }
  const int col  = (wave << 4) + (lane & 15);
  const int mb   = (lane >= 16) ? 8 : 0;
  const float bi = b_in[col], bo = b_out[col];
  const int bidx = seq0 >> 7;
  const int e0   = (seq0 & 127) + mb;
  #pragma unroll
  for (int r = 0; r < 8; ++r) {
    hin_t [(size_t)bidx * 16384 + (size_t)col * 128 + e0 + r] = f2bf(ci[r] + bi);
    hout_t[(size_t)bidx * 16384 + (size_t)col * 128 + e0 + r] = f2bf(co[r] + bo);
  }
}

// ---------------------------------------------------------------------------
// kernel 3: input_in/out = A[b,n,:128]@h_in[b] , A[b,n,128:]@h_out[b] + bias
// One block per (b, 16-row tile of N). B matrices come in pre-transposed.
// LDS: Hin 32KB | Hout 32KB | Ain 4KB | Aout 4KB = 72KB
// ---------------------------------------------------------------------------
__global__ __launch_bounds__(256) void einsum_kernel(
    const float* __restrict__ Ag, const bf16* __restrict__ hin_t,
    const bf16* __restrict__ hout_t, const float* __restrict__ b_iah,
    const float* __restrict__ b_ioh, bf16* __restrict__ inputs)
{
  extern __shared__ bf16 smem[];
  bf16* Hin  = smem;            // [d=128][e=128]
  bf16* Hout = Hin + 16384;
  bf16* Ain  = Hout + 16384;    // [16][128]
  bf16* Aout = Ain + 2048;

  const int tid  = threadIdx.x;
  const int lane = tid & 31;
  const int wave = tid >> 5;
  const int b    = blockIdx.x >> 3;
  const int n0   = (blockIdx.x & 7) << 4;

  {
    const int4* s0 = (const int4*)(hin_t  + (size_t)b * 16384);
    const int4* s1 = (const int4*)(hout_t + (size_t)b * 16384);
    int4* d0 = (int4*)Hin; int4* d1 = (int4*)Hout;
    for (int i = tid; i < 16384 / 8; i += 256) { d0[i] = s0[i]; d1[i] = s1[i]; }
  }
  for (int i = tid; i < 2048; i += 256) {
    int m = i >> 7, e = i & 127;
    size_t base = (size_t)(b * 128 + n0 + m) * 256;
    Ain[i]  = f2bf(Ag[base + e]);
    Aout[i] = f2bf(Ag[base + 128 + e]);
  }
  __syncthreads();

  v8f ci = zero8(), co = zero8();
  #pragma unroll
  for (int kk = 0; kk < 128; kk += 32) {
    ci = wmma_bf16(load_a_frag(Ain,  128, kk, lane),
                   load_b_frag(Hin,  128, wave << 4, kk, lane), ci);
    co = wmma_bf16(load_a_frag(Aout, 128, kk, lane),
                   load_b_frag(Hout, 128, wave << 4, kk, lane), co);
  }
  const int col  = (wave << 4) + (lane & 15);
  const int mb   = (lane >= 16) ? 8 : 0;
  const float bi = b_iah[col], bo = b_ioh[col];
  #pragma unroll
  for (int r = 0; r < 8; ++r) {
    size_t row = (size_t)(b * 128 + n0 + mb + r) * 256;
    inputs[row + col]       = f2bf(ci[r] + bi);
    inputs[row + 128 + col] = f2bf(co[r] + bo);
  }
}

// ---------------------------------------------------------------------------
// kernel 4: GNN gate. gi = inputs@w_ih.T, gh = hidden@w_hh.T, fused gates,
// f32 output. Each wave owns 6 accumulators like kernel 1.
// LDS: Wih 192KB | Whh 96KB | inputs 8KB | hidden 4KB = 300KB
// ---------------------------------------------------------------------------
__global__ __launch_bounds__(256) void gate_kernel(
    const bf16* __restrict__ inputs, const float* __restrict__ hidden,
    const bf16* __restrict__ wih_g, const bf16* __restrict__ whh_g,
    const float* __restrict__ bih, const float* __restrict__ bhh,
    float* __restrict__ out)
{
  extern __shared__ bf16 smem[];
  bf16* Wih = smem;             // [384][256]
  bf16* Whh = Wih + 98304;      // [384][128]
  bf16* Inp = Whh + 49152;      // [16][256]
  bf16* Hid = Inp + 4096;       // [16][128]

  const int tid  = threadIdx.x;
  const int lane = tid & 31;
  const int wave = tid >> 5;
  const int b    = blockIdx.x >> 3;
  const int n0   = (blockIdx.x & 7) << 4;
  const int row0 = b * 128 + n0;

  {
    const int4* s0 = (const int4*)wih_g; int4* d0 = (int4*)Wih;
    for (int i = tid; i < 98304 / 8; i += 256) d0[i] = s0[i];
    const int4* s1 = (const int4*)whh_g; int4* d1 = (int4*)Whh;
    for (int i = tid; i < 49152 / 8; i += 256) d1[i] = s1[i];
    const int4* s2 = (const int4*)(inputs + (size_t)row0 * 256); int4* d2 = (int4*)Inp;
    for (int i = tid; i < 4096 / 8; i += 256) d2[i] = s2[i];
  }
  for (int i = tid; i < 2048; i += 256) {
    int m = i >> 7, k = i & 127;
    Hid[i] = f2bf(hidden[(size_t)(row0 + m) * 128 + k]);
  }
  __syncthreads();

  v8f ir = zero8(), iz = zero8(), in_ = zero8();
  v8f hr = zero8(), hz = zero8(), hn = zero8();
  #pragma unroll
  for (int kk = 0; kk < 256; kk += 32) {
    v16bf a = load_a_frag(Inp, 256, kk, lane);
    ir  = wmma_bf16(a, load_b_frag(Wih, 256, (wave << 4),       kk, lane), ir);
    iz  = wmma_bf16(a, load_b_frag(Wih, 256, 128 + (wave << 4), kk, lane), iz);
    in_ = wmma_bf16(a, load_b_frag(Wih, 256, 256 + (wave << 4), kk, lane), in_);
  }
  #pragma unroll
  for (int kk = 0; kk < 128; kk += 32) {
    v16bf a = load_a_frag(Hid, 128, kk, lane);
    hr = wmma_bf16(a, load_b_frag(Whh, 128, (wave << 4),       kk, lane), hr);
    hz = wmma_bf16(a, load_b_frag(Whh, 128, 128 + (wave << 4), kk, lane), hz);
    hn = wmma_bf16(a, load_b_frag(Whh, 128, 256 + (wave << 4), kk, lane), hn);
  }

  const int col = (wave << 4) + (lane & 15);
  const int mb  = (lane >= 16) ? 8 : 0;
  const float bir = bih[col], biz = bih[128 + col], bin = bih[256 + col];
  const float bhr = bhh[col], bhz = bhh[128 + col], bhn = bhh[256 + col];
  #pragma unroll
  for (int r = 0; r < 8; ++r) {
    size_t idx = (size_t)(row0 + mb + r) * 128 + col;
    float hold = hidden[idx];
    float rg = fast_sigmoid(ir[r] + bir + hr[r] + bhr);
    float ig = fast_sigmoid(iz[r] + biz + hz[r] + bhz);
    float ng = fast_tanh(in_[r] + bin + rg * (hn[r] + bhn));
    out[idx] = (1.0f - ig) * hold + ig * ng;
  }
}

// ---------------------------------------------------------------------------
extern "C" void kernel_launch(void* const* d_in, const int* in_sizes, int n_in,
                              void* d_out, int out_size, void* d_ws, size_t ws_size,
                              hipStream_t stream) {
  const float* A             = (const float*)d_in[0];
  const float* hidden        = (const float*)d_in[1];
  const float* macro_items   = (const float*)d_in[2];
  const float* micro_actions = (const float*)d_in[3];
  const int*   micro_len     = (const int*)d_in[4];
  const float* gru_w_ih      = (const float*)d_in[5];
  const float* gru_w_hh      = (const float*)d_in[6];
  const float* gru_b_ih      = (const float*)d_in[7];
  const float* gru_b_hh      = (const float*)d_in[8];
  const float* w_ih          = (const float*)d_in[9];
  const float* w_hh          = (const float*)d_in[10];
  const float* b_ih          = (const float*)d_in[11];
  const float* b_hh          = (const float*)d_in[12];
  const float* b_iah         = (const float*)d_in[13];
  const float* b_ioh         = (const float*)d_in[14];
  const float* w_in          = (const float*)d_in[15];
  const float* b_in          = (const float*)d_in[16];
  const float* w_out         = (const float*)d_in[17];
  const float* b_out         = (const float*)d_in[18];
  float* out = (float*)d_out;

  // workspace partition (bf16 elements, all 16B aligned)
  bf16* mh     = (bf16*)d_ws;            // 32768*128
  bf16* hin_t  = mh     + 4194304;       // [B][D][E]
  bf16* hout_t = hin_t  + 4194304;
  bf16* inputs = hout_t + 4194304;       // [B][N][256]
  bf16* gwih   = inputs + 8388608;       // 384*128
  bf16* gwhh   = gwih   + 49152;
  bf16* wihb   = gwhh   + 49152;         // 384*256
  bf16* whhb   = wihb   + 98304;         // 384*128
  bf16* winb   = whhb   + 49152;         // 128*256
  bf16* woutb  = winb   + 32768;

  auto cvt = [&](const float* s, bf16* d, int n) {
    cvt_bf16_kernel<<<(n + 255) / 256, 256, 0, stream>>>(s, d, n);
  };
  cvt(gru_w_ih, gwih, 49152);
  cvt(gru_w_hh, gwhh, 49152);
  cvt(w_ih,     wihb, 98304);
  cvt(w_hh,     whhb, 49152);
  cvt(w_in,     winb, 32768);
  cvt(w_out,    woutb, 32768);

  gru_kernel<<<2048, 256, (size_t)(49152 * 2 + 2048 * 3) * sizeof(bf16), stream>>>(
      micro_actions, micro_len, gwih, gwhh, gru_b_ih, gru_b_hh, mh);

  inout_kernel<<<2048, 256, (size_t)(32768 * 2 + 4096) * sizeof(bf16), stream>>>(
      macro_items, mh, winb, woutb, b_in, b_out, hin_t, hout_t);

  einsum_kernel<<<2048, 256, (size_t)(16384 * 2 + 2048 * 2) * sizeof(bf16), stream>>>(
      A, hin_t, hout_t, b_iah, b_ioh, inputs);

  gate_kernel<<<2048, 256, (size_t)(98304 + 49152 + 4096 + 2048) * sizeof(bf16), stream>>>(
      inputs, hidden, wihb, whhb, b_ih, b_hh, out);
}
